// EquivariantWSSHead_84232898609308
// MI455X (gfx1250) — compile-verified, matrix operand order checked by reading, not run on hardware
//
#include <hip/hip_runtime.h>
#include <hip/hip_bf16.h>

typedef __attribute__((ext_vector_type(16))) _Float16 v16h;
typedef __attribute__((ext_vector_type(8)))  float    v8f;

// ---------------------------------------------------------------------------
// Weight-matrix entry (48 features x 16 output dots, col 15 unused/zero).
// Feature row layout matches x row: [0..15]=x0, [16+2c]=x1[...,0], [17+2c]=x1[...,1]
// Dot columns: 0:x0.w_n00 1:x0.w_n01[:,0] 2:x0.w_n01[:,1]
//              3..8 : x1s0 . {w_n10[:,0], w_n10[:,1], p, q, r, s}
//              9..14: x1s1 . {w_n10[:,0], w_n10[:,1], p, q, r, s}
// ---------------------------------------------------------------------------
__device__ inline float wv_entry(int f, int n,
                                 const float* w_n00, const float* w_n10,
                                 const float* w_n01, const float* w_n11) {
    if (f < 16) {
        if (n == 0) return w_n00[f];
        if (n == 1) return w_n01[2 * f];
        if (n == 2) return w_n01[2 * f + 1];
        return 0.0f;
    }
    if (f < 48) {
        int c    = (f - 16) >> 1;
        int comp = (f - 16) & 1;
        int j    = n - (comp ? 9 : 3);
        if (j < 0 || j > 5) return 0.0f;
        if (j == 0) return w_n10[2 * c];
        if (j == 1) return w_n10[2 * c + 1];
        return w_n11[4 * c + (j - 2)];           // p,q,r,s
    }
    return 0.0f;
}

__global__ void init_zero_kernel(float* p, int n) {
    int i = blockIdx.x * blockDim.x + threadIdx.x;
    if (i < n) p[i] = 0.0f;
}

// Pack B into the exact per-lane v16h layout of v_wmma_f32_16x16x32_f16.
// Lane l holds column n=l&15; element i maps to K = i + 8*(l>=16) + (i>=8 ? 8 : 0).
// Chunk0 covers features K (0..31), chunk1 covers features 32+K (valid only K<16).
__global__ void build_B_kernel(const float* w_n00, const float* w_n10,
                               const float* w_n01, const float* w_n11,
                               _Float16* B0, _Float16* B1) {
    int l = threadIdx.x;
    if (l >= 32) return;
    int n  = l & 15;
    int hi = l >> 4;
    for (int i = 0; i < 16; ++i) {
        int f = i + 8 * hi + (i >= 8 ? 8 : 0);
        B0[l * 16 + i] = (_Float16)wv_entry(f, n, w_n00, w_n10, w_n01, w_n11);
        B1[l * 16 + i] = (f < 16)
            ? (_Float16)wv_entry(32 + f, n, w_n00, w_n10, w_n01, w_n11)
            : (_Float16)0.0f;
    }
}

// Gather one A row (48 f32 features -> f16 WMMA lane layout, K padded to 64).
__device__ inline void gather_A(const float* __restrict__ row, int hi,
                                v16h& a0, v16h& a1) {
    const int fb0 = hi ? 8 : 0;          // first contiguous run of 8 features
    const int fb1 = hi ? 24 : 16;        // second contiguous run of 8 features
#pragma unroll
    for (int i = 0; i < 8; ++i) a0[i]     = (_Float16)row[fb0 + i];
#pragma unroll
    for (int i = 0; i < 8; ++i) a0[8 + i] = (_Float16)row[fb1 + i];
#pragma unroll
    for (int i = 0; i < 8; ++i) a1[i]     = (_Float16)row[32 + fb0 + i];
#pragma unroll
    for (int i = 0; i < 8; ++i) a1[8 + i] = (_Float16)0.0f;   // K=48..63 pad
}

// ---------------------------------------------------------------------------
// Edge kernel: one wave per 32-edge super-tile (2 x 16-edge WMMA tiles),
// 8 waves per block.  A (16x48 f16) x B (48x15 f16) -> D (16x16 f32), twice.
// All 32 lanes then own one edge each for the trig epilogue + atomic scatter.
// ---------------------------------------------------------------------------
__global__ __launch_bounds__(256)
void edge_kernel(const float* __restrict__ x,
                 const int*   __restrict__ src,
                 const int*   __restrict__ dst,
                 const float* __restrict__ angles,
                 const float* __restrict__ transporters,
                 const _Float16* __restrict__ B0p,
                 const _Float16* __restrict__ B1p,
                 float* __restrict__ deg,
                 float* __restrict__ magsum,
                 float* __restrict__ agg1,
                 float* __restrict__ agg2,
                 int E) {
    __shared__ float tile[8][2 * 16 * 16];

    const int lane = threadIdx.x & 31;
    const int wv   = threadIdx.x >> 5;
    const int m    = lane & 15;          // edge-in-tile (A row) / weight col (B)
    const int hi   = lane >> 4;

    const long tbase = ((long)blockIdx.x * 8 + wv) * 32;   // 32 edges per wave
    long e0 = tbase + m;                  // tile 0 edge for this lane's A row
    long e1 = tbase + 16 + m;             // tile 1 edge
    long e0c = e0 < (long)E ? e0 : (long)E - 1;
    long e1c = e1 < (long)E ? e1 : (long)E - 1;

    // edge-index loads: streamed once -> non-temporal so x stays L2-resident
    int s0 = __builtin_nontemporal_load(&src[e0c]);
    int s1 = __builtin_nontemporal_load(&src[e1c]);

    // ---- gather A rows into WMMA lane layout ----
    v16h a0t0, a1t0, a0t1, a1t1;
    gather_A(x + (size_t)s0 * 48, hi, a0t0, a1t0);
    gather_A(x + (size_t)s1 * 48, hi, a0t1, a1t1);

    // ---- B already packed in lane layout: one 32B load each ----
    v16h b0 = *(const v16h*)(B0p + lane * 16);
    v16h b1 = *(const v16h*)(B1p + lane * 16);

    v8f c0 = {}, c1 = {};
    c0 = __builtin_amdgcn_wmma_f32_16x16x32_f16(false, a0t0, false, b0,
                                                (short)0, c0, false, false);
    c0 = __builtin_amdgcn_wmma_f32_16x16x32_f16(false, a1t0, false, b1,
                                                (short)0, c0, false, false);
    c1 = __builtin_amdgcn_wmma_f32_16x16x32_f16(false, a0t1, false, b0,
                                                (short)0, c1, false, false);
    c1 = __builtin_amdgcn_wmma_f32_16x16x32_f16(false, a1t1, false, b1,
                                                (short)0, c1, false, false);

    // ---- D -> LDS so each epilogue lane owns one edge's 15 dots ----
    // D element (M=r+8*hi edge, N=lane&15 dot); tile1 rows at +256 floats.
    float* t = tile[wv];
#pragma unroll
    for (int r = 0; r < 8; ++r) {
        t[(r + 8 * hi) * 16 + m]       = c0[r];
        t[256 + (r + 8 * hi) * 16 + m] = c1[r];
    }
    __syncthreads();

    // ---- per-edge trig epilogue + scatter: lane l owns edge tbase+l ----
    // (tile1 row M at 256 + M*16 == l*16 for l>=16, so row addr is uniform)
    long ee = tbase + lane;
    if (ee < (long)E) {
        const float* d = &t[lane * 16];
        float ang = __builtin_nontemporal_load(&angles[ee]);
        float tr  = __builtin_nontemporal_load(&transporters[ee]);
        float st, ct, s2t, c2t, sg, cg;
        __sincosf(ang, &st, &ct);
        __sincosf(2.0f * ang, &s2t, &c2t);
        __sincosf(tr, &sg, &cg);

        float d0 = d[0], sa = d[1], sb = d[2];
        // rotate raw dots by transporter: u1.w = cg*d_s0 - sg*d_s1 ; u2.w = sg*d_s0 + cg*d_s1
        float u1w0 = cg * d[3] - sg * d[9];
        float u2w0 = sg * d[3] + cg * d[9];
        float u1w1 = cg * d[4] - sg * d[10];
        float u2w1 = sg * d[4] + cg * d[10];
        float u1p  = cg * d[5] - sg * d[11];
        float u2p  = sg * d[5] + cg * d[11];
        float u1q  = cg * d[6] - sg * d[12];
        float u2q  = sg * d[6] + cg * d[12];
        float u1r  = cg * d[7] - sg * d[13];
        float u2r  = sg * d[7] + cg * d[13];
        float u1s  = cg * d[8] - sg * d[14];
        float u2s  = sg * d[8] + cg * d[14];

        float m0  = d0 + ct * u1w0 + st * u2w0 + ct * u2w1 - st * u1w1;
        float mv1 = sa * ct - sb * st + u1p - u2q
                  + (c2t * u1r + s2t * u2r) - (s2t * u1s - c2t * u2s);
        float mv2 = sa * st + sb * ct + u2p + u1q
                  + (s2t * u1r - c2t * u2r) + (c2t * u1s + s2t * u2s);

        int dv = __builtin_nontemporal_load(&dst[ee]);
        __hip_atomic_fetch_add(&deg[dv],    1.0f, __ATOMIC_RELAXED, __HIP_MEMORY_SCOPE_AGENT);
        __hip_atomic_fetch_add(&magsum[dv], m0,   __ATOMIC_RELAXED, __HIP_MEMORY_SCOPE_AGENT);
        __hip_atomic_fetch_add(&agg1[dv],   mv1,  __ATOMIC_RELAXED, __HIP_MEMORY_SCOPE_AGENT);
        __hip_atomic_fetch_add(&agg2[dv],   mv2,  __ATOMIC_RELAXED, __HIP_MEMORY_SCOPE_AGENT);
    }
}

// ---------------------------------------------------------------------------
// Per-vertex finalize: self terms, sigmoid gate, project onto e1/e2.
// ---------------------------------------------------------------------------
__global__ __launch_bounds__(256)
void finalize_kernel(const float* __restrict__ x,
                     const float* __restrict__ e1,
                     const float* __restrict__ e2,
                     const float* __restrict__ w_self0,
                     const float* __restrict__ w_self11,
                     const float* __restrict__ deg,
                     const float* __restrict__ magsum,
                     const float* __restrict__ agg1,
                     const float* __restrict__ agg2,
                     float* __restrict__ out,
                     int V) {
    int v = blockIdx.x * blockDim.x + threadIdx.x;
    if (v >= V) return;
    const float* row = x + (size_t)v * 48;
    float denom = fmaxf(deg[v], 1.0f);
    float inv   = 1.0f / denom;
    float mag   = magsum[v] * inv;
    float t1    = agg1[v] * inv;
    float t2    = agg2[v] * inv;
#pragma unroll
    for (int c = 0; c < 16; ++c) {
        mag += row[c] * w_self0[c];
        float xa = row[16 + 2 * c], xb = row[17 + 2 * c];
        float sa = w_self11[2 * c], sb = w_self11[2 * c + 1];
        t1 += xa * sa - xb * sb;
        t2 += xb * sa + xa * sb;
    }
    float ms = 2.0f / (1.0f + __expf(-mag));
#pragma unroll
    for (int k = 0; k < 3; ++k)
        out[v * 3 + k] = (t1 * e1[v * 3 + k] + t2 * e2[v * 3 + k]) * ms;
}

// ---------------------------------------------------------------------------
extern "C" void kernel_launch(void* const* d_in, const int* in_sizes, int n_in,
                              void* d_out, int out_size, void* d_ws, size_t ws_size,
                              hipStream_t stream) {
    const float* x            = (const float*)d_in[0];
    const int*   edge_index   = (const int*)d_in[1];    // (2,E) int32
    const float* angles       = (const float*)d_in[2];
    const float* transporters = (const float*)d_in[3];
    const float* e1           = (const float*)d_in[4];
    const float* e2           = (const float*)d_in[5];
    const float* w_self0      = (const float*)d_in[6];
    const float* w_n00        = (const float*)d_in[7];
    const float* w_n10        = (const float*)d_in[8];
    const float* w_self11     = (const float*)d_in[9];
    const float* w_n01        = (const float*)d_in[10];
    const float* w_n11        = (const float*)d_in[11];

    const int E = in_sizes[2];           // angles: (E,)
    const int V = in_sizes[0] / 48;      // x: (V, 48)
    const int* src = edge_index;
    const int* dst = edge_index + E;

    // workspace layout: 4*V f32 accumulators, then two packed B operands
    float* deg    = (float*)d_ws;
    float* magsum = deg + V;
    float* agg1   = magsum + V;
    float* agg2   = agg1 + V;
    _Float16* B0  = (_Float16*)(agg2 + V);
    _Float16* B1  = B0 + 32 * 16;

    int nz = 4 * V;
    hipLaunchKernelGGL(init_zero_kernel, dim3((nz + 255) / 256), dim3(256), 0, stream,
                       deg, nz);
    hipLaunchKernelGGL(build_B_kernel, dim3(1), dim3(32), 0, stream,
                       w_n00, w_n10, w_n01, w_n11, B0, B1);

    int stiles = (E + 31) / 32;          // 32-edge super-tiles (1 wave each)
    int blocks = (stiles + 7) / 8;       // 8 waves per 256-thread block
    hipLaunchKernelGGL(edge_kernel, dim3(blocks), dim3(256), 0, stream,
                       x, src, dst, angles, transporters, B0, B1,
                       deg, magsum, agg1, agg2, E);

    hipLaunchKernelGGL(finalize_kernel, dim3((V + 255) / 256), dim3(256), 0, stream,
                       x, e1, e2, w_self0, w_self11, deg, magsum, agg1, agg2,
                       (float*)d_out, V);
}
